// Block_65094524338511
// MI455X (gfx1250) — compile-verified, matrix operand order checked by reading, not run on hardware
//
#include <hip/hip_runtime.h>
#include <stdint.h>

// ---------------------------------------------------------------------------
// CDNA5 (gfx1250) transformer block: 2x MHA + FFN.
// - One-time weight transpose+convert to bf16 [N,K]; activations carry bf16
//   shadow copies written by producer epilogues.
// - GEMMs: 128x128 block tile, BK=64, 8 waves x (64x32) wave tiles,
//   double-buffered LDS fed by global_load_async_to_lds_b128 (ASYNCcnt
//   pipelining), v_wmma_f32_16x16x32_bf16 compute (16 WMMA / stage / wave).
// - Attention core (0.4% of FLOPs) on VALU, one wave32 per (b,h).
// ---------------------------------------------------------------------------

typedef __attribute__((ext_vector_type(16))) __bf16 v16bf;
typedef __attribute__((ext_vector_type(8)))  __bf16 v8bf;
typedef __attribute__((ext_vector_type(8)))  float  v8f;

#define AST 72                     // bf16 elems per LDS tile row (144B = 9*16B)
#define STAGE_ELEMS (128 * AST)    // one 128x64 tile (padded)
#define STAGE_BYTES (128 * 144)

__device__ inline v16bf frag16(const __bf16* p) {
  v8bf lo = *(const v8bf*)p;          // ds_load_b128
  v8bf hi = *(const v8bf*)(p + 8);    // ds_load_b128
  v16bf r;
#pragma unroll
  for (int i = 0; i < 8; ++i) { r[i] = lo[i]; r[i + 8] = hi[i]; }
  return r;
}

// out(fp32, optional) address = (m>>5)*s0 + (m&31)*s1 + (n>>6)*s2 + (n&nmask)
// outb(bf16, optional) address = m*ldob + n   (plain row-major, feeds next GEMM)
template <bool RELU>
__global__ __launch_bounds__(256)
void gemm_bf16_wmma(const __bf16* __restrict__ A,      // [M,K] bf16 row-major
                    const __bf16* __restrict__ Bt,     // [N,K] bf16 row-major
                    float* __restrict__ out, int64_t s0, int64_t s1, int64_t s2,
                    int64_t nmask,
                    __bf16* __restrict__ outb, int64_t ldob,
                    const float* __restrict__ bias,
                    const float* __restrict__ resid, int64_t ldres,
                    int K)
{
  __shared__ __align__(16) __bf16 lds[2 * 2 * STAGE_ELEMS]; // [A0][B0][A1][B1]

  const int tid   = threadIdx.x;
  const int lane  = tid & 31;
  const int wave  = tid >> 5;
  const int waveM = wave >> 2;          // 0..1 : 64-row slab
  const int waveN = wave & 3;           // 0..3 : 32-col slab
  const int64_t blockM = (int64_t)blockIdx.y * 128;
  const int64_t blockN = (int64_t)blockIdx.x * 128;
  const int lrow = lane & 15;
  const int hseg = (lane >> 4) * 16;    // K sub-segment (bf16 elems)

  // --- async staging addresses: each thread copies 4x16B of A and of B ---
  const int cpRow = tid >> 1;           // 0..127
  const int cpSeg = (tid & 1) * 4;      // 16B segments 0..3 or 4..7
  const char* gA = (const char*)(A  + (blockM + cpRow) * (int64_t)K + cpSeg * 8);
  const char* gB = (const char*)(Bt + (blockN + cpRow) * (int64_t)K + cpSeg * 8);
  const uint32_t ldsBase = (uint32_t)(uintptr_t)(&lds[0]);
  const uint32_t lA = ldsBase + cpRow * 144 + cpSeg * 16;
  const uint32_t lB = lA + STAGE_BYTES;

  v8f acc[4][2];
#pragma unroll
  for (int mi = 0; mi < 4; ++mi)
#pragma unroll
    for (int nj = 0; nj < 2; ++nj)
#pragma unroll
      for (int r = 0; r < 8; ++r) acc[mi][nj][r] = 0.0f;

  auto prefetch = [&](int buf, int k0) {
    const char* a = gA + (int64_t)k0 * 2;
    const char* b = gB + (int64_t)k0 * 2;
    const uint32_t la = lA + buf * (2 * STAGE_BYTES);
    const uint32_t lb = lB + buf * (2 * STAGE_BYTES);
#pragma unroll
    for (int i = 0; i < 4; ++i) {
      asm volatile("global_load_async_to_lds_b128 %0, %1, off"
                   :: "v"(la + i * 16), "v"(a + i * 16) : "memory");
      asm volatile("global_load_async_to_lds_b128 %0, %1, off"
                   :: "v"(lb + i * 16), "v"(b + i * 16) : "memory");
    }
  };

  prefetch(0, 0);                                    // ASYNCcnt = 8
  int buf = 0;
  for (int k0 = 0; k0 < K; k0 += 64) {
    if (k0 + 64 < K) {
      prefetch(buf ^ 1, k0 + 64);                    // ASYNCcnt -> 16
      asm volatile("s_wait_asynccnt 8" ::: "memory"); // stage `buf` landed
    } else {
      asm volatile("s_wait_asynccnt 0" ::: "memory");
    }
    __syncthreads();                                  // all waves' copies done

    const int aBase = buf * (2 * STAGE_ELEMS);
    const int bBase = aBase + STAGE_ELEMS;
#pragma unroll
    for (int ks = 0; ks < 2; ++ks) {
      v16bf af[4], bfr[2];
#pragma unroll
      for (int mi = 0; mi < 4; ++mi)
        af[mi] = frag16(&lds[aBase + (waveM * 64 + mi * 16 + lrow) * AST + ks * 32 + hseg]);
#pragma unroll
      for (int nj = 0; nj < 2; ++nj)
        bfr[nj] = frag16(&lds[bBase + (waveN * 32 + nj * 16 + lrow) * AST + ks * 32 + hseg]);
#pragma unroll
      for (int mi = 0; mi < 4; ++mi)
#pragma unroll
        for (int nj = 0; nj < 2; ++nj)
          acc[mi][nj] = __builtin_amdgcn_wmma_f32_16x16x32_bf16(
              false, af[mi], false, bfr[nj], (short)0, acc[mi][nj], false, false);
    }
    __syncthreads();            // tile fully consumed before it is re-filled
    buf ^= 1;
  }

  // --- epilogue: bias (+resid) (relu), fp32 scatter and/or bf16 copy ---
#pragma unroll
  for (int mi = 0; mi < 4; ++mi)
#pragma unroll
    for (int nj = 0; nj < 2; ++nj) {
      int64_t n = blockN + waveN * 32 + nj * 16 + lrow;
      float bv = bias ? bias[n] : 0.0f;
#pragma unroll
      for (int r = 0; r < 8; ++r) {
        int64_t m = blockM + waveM * 64 + mi * 16 + r + ((lane >> 4) << 3);
        float v = acc[mi][nj][r] + bv;
        if (resid) v += resid[m * ldres + n];
        if (RELU) v = fmaxf(v, 0.0f);
        if (out)  out[(m >> 5) * s0 + (m & 31) * s1 + (n >> 6) * s2 + (n & nmask)] = v;
        if (outb) outb[m * ldob + n] = (__bf16)v;
      }
    }
}

// One-time weight transpose + fp32->bf16: W(src, head-mapped) -> Wt [N,K] bf16.
// Source element (k,n) at W + (n>>nshift)*zs + k*ldw + (n&nmask).
__global__ __launch_bounds__(256)
void transpose_w_bf16(const float* __restrict__ W, int ldw, int64_t zs,
                      int nshift, int nmask,
                      __bf16* __restrict__ Wt, int K)
{
  __shared__ float tile[32][33];
  const int k0 = blockIdx.x * 32, n0 = blockIdx.y * 32;
  const int tx = threadIdx.x, ty = threadIdx.y;   // 32 x 8
#pragma unroll
  for (int i = 0; i < 32; i += 8) {
    int k = k0 + ty + i, n = n0 + tx;
    tile[ty + i][tx] = W[(int64_t)(n >> nshift) * zs + (int64_t)k * ldw + (n & nmask)];
  }
  __syncthreads();
#pragma unroll
  for (int i = 0; i < 32; i += 8) {
    int n = n0 + ty + i, k = k0 + tx;
    Wt[(int64_t)n * K + k] = (__bf16)tile[tx][ty + i];
  }
}

__global__ __launch_bounds__(256)
void f32_to_bf16(const float* __restrict__ src, __bf16* __restrict__ dst)
{
  int64_t i = ((int64_t)blockIdx.x * 256 + threadIdx.x) * 4;
  float4 f = *(const float4*)(src + i);
  dst[i + 0] = (__bf16)f.x; dst[i + 1] = (__bf16)f.y;
  dst[i + 2] = (__bf16)f.z; dst[i + 3] = (__bf16)f.w;
}

// Attention core: one wave32 per (b,h). q/k/v: [B,H,32,64] fp32.
// out: bf16 [B*T, H*64] (head-concatenated, feeds Wp GEMM directly).
#define QSTR 68
__global__ __launch_bounds__(32)
void attention_kernel(const float* __restrict__ q, const float* __restrict__ k,
                      const float* __restrict__ v, __bf16* __restrict__ out, int H)
{
  __shared__ __align__(16) float Qs[32 * QSTR];
  __shared__ __align__(16) float Ks[32 * QSTR];
  __shared__ __align__(16) float Vs[32 * QSTR];

  const int bh = blockIdx.x;
  const int b  = bh / H;
  const int h  = bh % H;
  const int t  = threadIdx.x;
  const int64_t base = (int64_t)bh * 32 * 64;

#pragma unroll
  for (int i = 0; i < 16; ++i) {
    *(float4*)&Qs[t * QSTR + i * 4] = *(const float4*)(q + base + t * 64 + i * 4);
    *(float4*)&Ks[t * QSTR + i * 4] = *(const float4*)(k + base + t * 64 + i * 4);
    *(float4*)&Vs[t * QSTR + i * 4] = *(const float4*)(v + base + t * 64 + i * 4);
  }
  __syncthreads();

  float sc[32];
#pragma unroll
  for (int s = 0; s < 32; ++s) {
    float d = 0.0f;
#pragma unroll
    for (int e = 0; e < 64; ++e) d += Qs[t * QSTR + e] * Ks[s * QSTR + e];
    sc[s] = d * 0.125f;
  }
  float mx = -1e30f;
#pragma unroll
  for (int s = 0; s < 32; ++s) { if (s > t) sc[s] = -1e30f; mx = fmaxf(mx, sc[s]); }
  float sum = 0.0f;
#pragma unroll
  for (int s = 0; s < 32; ++s) { sc[s] = __expf(sc[s] - mx); sum += sc[s]; }
  const float inv = 1.0f / sum;

  const int64_t obase = ((int64_t)b * 32 + t) * (int64_t)(H * 64) + h * 64;
#pragma unroll
  for (int e = 0; e < 64; ++e) {
    float o = 0.0f;
#pragma unroll
    for (int s = 0; s < 32; ++s) o += sc[s] * Vs[s * QSTR + e];
    out[obase + e] = (__bf16)(o * inv);
  }
}

// ---------------------------------------------------------------------------
// Host side
// ---------------------------------------------------------------------------
static inline void gemm(const __bf16* A, const __bf16* Bt,
                        float* out, int64_t s0, int64_t s1, int64_t s2, int64_t nmask,
                        __bf16* outb, int64_t ldob,
                        const float* bias, const float* resid, int64_t ldres,
                        int M, int N, int K, bool relu, hipStream_t st)
{
  dim3 g(N / 128, M / 128), blk(256);
  if (relu)
    gemm_bf16_wmma<true><<<g, blk, 0, st>>>(A, Bt, out, s0, s1, s2, nmask,
                                            outb, ldob, bias, resid, ldres, K);
  else
    gemm_bf16_wmma<false><<<g, blk, 0, st>>>(A, Bt, out, s0, s1, s2, nmask,
                                             outb, ldob, bias, resid, ldres, K);
}

extern "C" void kernel_launch(void* const* d_in, const int* in_sizes, int n_in,
                              void* d_out, int out_size, void* d_ws, size_t ws_size,
                              hipStream_t stream) {
  (void)in_sizes; (void)n_in; (void)out_size; (void)ws_size;

  const float* x    = (const float*)d_in[0];
  const float* Wq1  = (const float*)d_in[1];
  const float* bq1  = (const float*)d_in[2];
  const float* Wk1  = (const float*)d_in[3];
  const float* bk1  = (const float*)d_in[4];
  const float* Wv1  = (const float*)d_in[5];
  const float* bv1  = (const float*)d_in[6];
  const float* Wp1  = (const float*)d_in[7];
  const float* bp1  = (const float*)d_in[8];
  const float* Wq2  = (const float*)d_in[9];
  const float* bq2  = (const float*)d_in[10];
  const float* Wk2  = (const float*)d_in[11];
  const float* bk2  = (const float*)d_in[12];
  const float* Wv2  = (const float*)d_in[13];
  const float* bv2  = (const float*)d_in[14];
  const float* Wp2  = (const float*)d_in[15];
  const float* bp2  = (const float*)d_in[16];
  const float* Wff1 = (const float*)d_in[17];
  const float* bff1 = (const float*)d_in[18];
  const float* Wff2 = (const float*)d_in[19];
  const float* bff2 = (const float*)d_in[20];

  const int B = 1024, E = 1024, H = 16, D = 64, Fh = 4096;
  const int M = B * 32;                                   // 32768
  char* ws = (char*)d_ws;

  // ---- workspace layout (byte offsets) ----
  __bf16* xb    = (__bf16*)(ws + 0);                      //  64MB bf16 [M,E]
  char*   wbase = ws + (int64_t)67108864;
  __bf16* Wq1t  = (__bf16*)(wbase + 0LL * 2097152);       // bf16 [1024,1024] each
  __bf16* Wk1t  = (__bf16*)(wbase + 1LL * 2097152);
  __bf16* Wv1t  = (__bf16*)(wbase + 2LL * 2097152);
  __bf16* Wp1t  = (__bf16*)(wbase + 3LL * 2097152);
  __bf16* Wq2t  = (__bf16*)(wbase + 4LL * 2097152);
  __bf16* Wk2t  = (__bf16*)(wbase + 5LL * 2097152);
  __bf16* Wv2t  = (__bf16*)(wbase + 6LL * 2097152);
  __bf16* Wp2t  = (__bf16*)(wbase + 7LL * 2097152);
  __bf16* Wff1t = (__bf16*)(wbase + 8LL * 2097152);       // bf16 [4096,1024]
  __bf16* Wff2t = (__bf16*)(wbase + 12LL * 2097152);      // bf16 [1024,4096]
  float*  qb    = (float*) (ws + (int64_t)100663296);     // 128MB fp32 [B,H,32,64]
  float*  kb    = (float*) (ws + (int64_t)234881024);
  float*  vb    = (float*) (ws + (int64_t)369098752);
  __bf16* atb   = (__bf16*)(ws + (int64_t)503316480);     //  64MB bf16 [M,E]
  float*  x1    = (float*) (ws + (int64_t)570425344);     // 128MB fp32 [M,E]
  __bf16* x1b   = (__bf16*)(ws + (int64_t)704643072);     //  64MB bf16 [M,E]
  float*  x2    = qb;                                     // reuse (q dead)
  __bf16* x2b   = (__bf16*)kb;                            // reuse (k dead)
  __bf16* hb    = (__bf16*)vb;                            // 256MB, spans v/atb/x1 (dead)
  float*  y     = (float*)d_out;

  const int64_t qs0 = (int64_t)H * 32 * D;  // 32768  batch stride of [B,H,T,D]
  const int64_t qs2 = (int64_t)32 * D;      // 2048   head stride
  const int64_t ALL = 0x7FFFFFFFLL;         // nmask for plain outputs (s2=0)

  // ---- one-time weight transpose+convert (reused 256x by GEMMs) ----
  dim3 tb(32, 8);
  transpose_w_bf16<<<dim3(32, 32),  tb, 0, stream>>>(Wq1,  D, (int64_t)E * D, 6, 63, Wq1t, E);
  transpose_w_bf16<<<dim3(32, 32),  tb, 0, stream>>>(Wk1,  D, (int64_t)E * D, 6, 63, Wk1t, E);
  transpose_w_bf16<<<dim3(32, 32),  tb, 0, stream>>>(Wv1,  D, (int64_t)E * D, 6, 63, Wv1t, E);
  transpose_w_bf16<<<dim3(32, 32),  tb, 0, stream>>>(Wp1,  E, 0, 30, 0x7FFFFFFF, Wp1t, E);
  transpose_w_bf16<<<dim3(32, 32),  tb, 0, stream>>>(Wq2,  D, (int64_t)E * D, 6, 63, Wq2t, E);
  transpose_w_bf16<<<dim3(32, 32),  tb, 0, stream>>>(Wk2,  D, (int64_t)E * D, 6, 63, Wk2t, E);
  transpose_w_bf16<<<dim3(32, 32),  tb, 0, stream>>>(Wv2,  D, (int64_t)E * D, 6, 63, Wv2t, E);
  transpose_w_bf16<<<dim3(32, 32),  tb, 0, stream>>>(Wp2,  E, 0, 30, 0x7FFFFFFF, Wp2t, E);
  transpose_w_bf16<<<dim3(32, 128), tb, 0, stream>>>(Wff1, Fh, 0, 30, 0x7FFFFFFF, Wff1t, E);
  transpose_w_bf16<<<dim3(128, 32), tb, 0, stream>>>(Wff2, E, 0, 30, 0x7FFFFFFF, Wff2t, Fh);
  f32_to_bf16<<<dim3(32768), dim3(256), 0, stream>>>(x, xb);

  // ---- layer 1 ----
  gemm(xb, Wq1t, qb, qs0, D, qs2, 63, nullptr, 0, bq1, nullptr, 0, M, E, E, false, stream);
  gemm(xb, Wk1t, kb, qs0, D, qs2, 63, nullptr, 0, bk1, nullptr, 0, M, E, E, false, stream);
  gemm(xb, Wv1t, vb, qs0, D, qs2, 63, nullptr, 0, bv1, nullptr, 0, M, E, E, false, stream);
  attention_kernel<<<dim3(B * H), dim3(32), 0, stream>>>(qb, kb, vb, atb, H);
  gemm(atb, Wp1t, x1, (int64_t)32 * E, E, 0, ALL, x1b, E, bp1, x, E, M, E, E, false, stream);

  // ---- layer 2 ----
  gemm(x1b, Wq2t, qb, qs0, D, qs2, 63, nullptr, 0, bq2, nullptr, 0, M, E, E, false, stream);
  gemm(x1b, Wk2t, kb, qs0, D, qs2, 63, nullptr, 0, bk2, nullptr, 0, M, E, E, false, stream);
  gemm(x1b, Wv2t, vb, qs0, D, qs2, 63, nullptr, 0, bv2, nullptr, 0, M, E, E, false, stream);
  attention_kernel<<<dim3(B * H), dim3(32), 0, stream>>>(qb, kb, vb, atb, H);
  gemm(atb, Wp2t, x2, (int64_t)32 * E, E, 0, ALL, x2b, E, bp2, x1, E, M, E, E, false, stream);

  // ---- FFN ----
  gemm(x2b, Wff1t, nullptr, 0, 0, 0, ALL, hb, Fh, bff1, nullptr, 0, M, Fh, E, true, stream);
  gemm(hb, Wff2t, y, (int64_t)32 * E, E, 0, ALL, nullptr, 0, bff2, x2, E, M, E, Fh, false, stream);
}
